// GTLayer_30545807409988
// MI455X (gfx1250) — matrix operand order repaired and checked by reading, not compiled, not run on hardware
//
#include <hip/hip_runtime.h>

typedef __attribute__((ext_vector_type(8)))  float   v8f;
typedef __attribute__((ext_vector_type(16))) __bf16  v16bf;
typedef __attribute__((ext_vector_type(8)))  __bf16  v8bf;
typedef __attribute__((ext_vector_type(4)))  __bf16  v4bf;

#define NROWS   8192
#define DIM     512
#define HID     1024
#define MROWS   32     // rows per block (2 WMMA row-tiles)
#define XPITCH  520    // 512 + 8 halves pad: row stride 1040B (16B aligned, 4-bank skew)
#define YPITCH  1032   // 1024 + 8 halves pad: row stride 2064B

// ---------------- column sum of h: csum[d] = sum_n h[n,d] ----------------
__global__ __launch_bounds__(256) void colsum_kernel(const float* __restrict__ h,
                                                     float* __restrict__ csum) {
    int t = threadIdx.x;
    int rbase = blockIdx.x * 128;
    float s0 = 0.f, s1 = 0.f;
    for (int r = 0; r < 128; ++r) {
        const float* row = h + (size_t)(rbase + r) * DIM;
        s0 += row[t];
        s1 += row[t + 256];
    }
    atomicAdd(&csum[t], s0);
    atomicAdd(&csum[t + 256], s1);
}

// ------------- transpose+convert weights: dst[n*K+k] = bf16(src[k*Nw+n]) -------------
__global__ __launch_bounds__(256) void prep_w_kernel(const float* __restrict__ src,
                                                     __bf16* __restrict__ dst,
                                                     int Nw, int K) {
    int i = blockIdx.x * 256 + threadIdx.x;      // total Nw*K = 524288 either way
    if (i < Nw * K) {
        int k = i % K;
        int n = i / K;
        dst[i] = (__bf16)src[(size_t)k * Nw + n];
    }
}

// ---------- fold attention (=colsum broadcast), ow/ob, residual, BN1, BN2 ----------
// cvec layout: a1[512] | c1[512] | a2[512] | cc2[512]
__global__ __launch_bounds__(512) void stats_kernel(
    const float* __restrict__ csum,
    const float* __restrict__ vw, const float* __restrict__ vb,
    const float* __restrict__ ow, const float* __restrict__ ob,
    const float* __restrict__ f2b,
    const float* __restrict__ g1, const float* __restrict__ b1,
    const float* __restrict__ m1, const float* __restrict__ v1,
    const float* __restrict__ g2, const float* __restrict__ b2,
    const float* __restrict__ m2, const float* __restrict__ v2,
    float* __restrict__ cvec) {
    __shared__ float sh[DIM];
    int d = threadIdx.x;
    sh[d] = csum[d];
    __syncthreads();
    // s = colsum(h) @ vw + N * vb   (== colsum of v; attn is all-ones since H==1)
    float s = 8192.0f * vb[d];
    for (int k = 0; k < DIM; ++k) s += sh[k] * vw[k * DIM + d];
    __syncthreads();
    sh[d] = s;
    __syncthreads();
    // t = s @ ow + ob  (row-constant part of x before residual/BN1)
    float t = ob[d];
    for (int k = 0; k < DIM; ++k) t += sh[k] * ow[k * DIM + d];
    float a1 = g1[d] * rsqrtf(v1[d] + 1e-5f);
    float c1 = (t - m1[d]) * a1 + b1[d];           // x = a1*h + c1
    float a2 = g2[d] * rsqrtf(v2[d] + 1e-5f);
    float cc2 = (f2b[d] - m2[d]) * a2 + b2[d];     // out = (x + acc)*a2 + cc2
    cvec[d]          = a1;
    cvec[d + 512]    = c1;
    cvec[d + 1024]   = a2;
    cvec[d + 1536]   = cc2;
}

// ------------------------- fused FFN with bf16 WMMA -------------------------
// Each block: 32 rows. Each wave: per column tile, two row-tiles share one B fragment.
__global__ __launch_bounds__(256) void ffn_kernel(
    const float* __restrict__ h,
    const float* __restrict__ f1b,
    const float* __restrict__ cvec,
    const __bf16* __restrict__ f1wT,   // [HID][DIM] col-major (contig in k)
    const __bf16* __restrict__ f2wT,   // [DIM][HID] col-major (contig in k)
    float* __restrict__ out) {
    __shared__ __align__(16) __bf16 xb[MROWS * XPITCH];   // ~33 KB
    __shared__ __align__(16) __bf16 yb[MROWS * YPITCH];   // ~66 KB

    const float* a1  = cvec;
    const float* c1  = cvec + 512;
    const float* a2  = cvec + 1024;
    const float* cc2 = cvec + 1536;

    const int r0    = blockIdx.x * MROWS;
    const int tid   = threadIdx.x;
    const int wave  = tid >> 5;
    const int lane  = tid & 31;
    const int lhalf = lane >> 4;     // 0: lanes 0-15, 1: lanes 16-31
    const int l15   = lane & 15;

    // Phase 1: x = a1*h + c1, vectorized: float4 loads -> packed 4x bf16 LDS stores
    for (int i = 0; i < 16; ++i) {
        int idx = tid + i * 256;            // 4096 float4 groups total
        int m  = idx >> 7;                  // 128 groups per row
        int d4 = (idx & 127) << 2;
        float4 hv = *(const float4*)&h[(size_t)(r0 + m) * DIM + d4];
        float4 va = *(const float4*)&a1[d4];
        float4 vc = *(const float4*)&c1[d4];
        v4bf pk;
        pk[0] = (__bf16)(va.x * hv.x + vc.x);
        pk[1] = (__bf16)(va.y * hv.y + vc.y);
        pk[2] = (__bf16)(va.z * hv.z + vc.z);
        pk[3] = (__bf16)(va.w * hv.w + vc.w);
        *(v4bf*)&xb[m * XPITCH + d4] = pk;
    }
    __syncthreads();

    // Phase 2: y = relu(x @ f1w + f1b)  -> bf16 LDS tile [32][1024]
    // A frag (16-bit 16x32): lane half 0 holds K {0..7,16..23}, half 1 holds K {8..15,24..31}
    for (int ct = 0; ct < 8; ++ct) {
        int nb = (wave * 8 + ct) * 16;
        int n  = nb + l15;
        v8f acc0 = {}, acc1 = {};
        const __bf16* ap0 = xb + l15 * XPITCH + lhalf * 8;          // rows 0-15
        const __bf16* ap1 = xb + (16 + l15) * XPITCH + lhalf * 8;   // rows 16-31
        const __bf16* bp  = f1wT + (size_t)n * DIM + lhalf * 8;
        #pragma unroll 4
        for (int ks = 0; ks < DIM; ks += 32) {
            union { v16bf v; v8bf q[2]; } A0, A1, B;
            B.q[0]  = *(const v8bf*)(bp + ks);        // K = ks + lhalf*8 + 0..7
            B.q[1]  = *(const v8bf*)(bp + ks + 16);   // K = ks + lhalf*8 + 16..23
            A0.q[0] = *(const v8bf*)(ap0 + ks);       // K = ks + kb + 0..7
            A0.q[1] = *(const v8bf*)(ap0 + ks + 16);  // K = ks + kb + 16..23
            A1.q[0] = *(const v8bf*)(ap1 + ks);
            A1.q[1] = *(const v8bf*)(ap1 + ks + 16);
            acc0 = __builtin_amdgcn_wmma_f32_16x16x32_bf16(
                false, A0.v, false, B.v, (short)0, acc0, false, false);
            acc1 = __builtin_amdgcn_wmma_f32_16x16x32_bf16(
                false, A1.v, false, B.v, (short)0, acc1, false, false);
        }
        float bias = f1b[n];
        for (int r = 0; r < 8; ++r) {
            int m = lhalf ? (r + 8) : r;              // C/D layout
            float y0 = acc0[r] + bias; y0 = y0 > 0.f ? y0 : 0.f;
            float y1 = acc1[r] + bias; y1 = y1 > 0.f ? y1 : 0.f;
            yb[m * YPITCH + n]        = (__bf16)y0;
            yb[(16 + m) * YPITCH + n] = (__bf16)y1;
        }
    }
    __syncthreads();

    // Phase 3: f = y @ f2w (+f2b folded into cc2); out = (x + f)*a2 + cc2
    for (int ct = 0; ct < 4; ++ct) {
        int nb = (wave * 4 + ct) * 16;
        int n  = nb + l15;
        v8f acc0 = {}, acc1 = {};
        const __bf16* ap0 = yb + l15 * YPITCH + lhalf * 8;
        const __bf16* ap1 = yb + (16 + l15) * YPITCH + lhalf * 8;
        const __bf16* bp  = f2wT + (size_t)n * HID + lhalf * 8;
        #pragma unroll 4
        for (int ks = 0; ks < HID; ks += 32) {
            union { v16bf v; v8bf q[2]; } A0, A1, B;
            B.q[0]  = *(const v8bf*)(bp + ks);
            B.q[1]  = *(const v8bf*)(bp + ks + 16);
            A0.q[0] = *(const v8bf*)(ap0 + ks);
            A0.q[1] = *(const v8bf*)(ap0 + ks + 16);
            A1.q[0] = *(const v8bf*)(ap1 + ks);
            A1.q[1] = *(const v8bf*)(ap1 + ks + 16);
            acc0 = __builtin_amdgcn_wmma_f32_16x16x32_bf16(
                false, A0.v, false, B.v, (short)0, acc0, false, false);
            acc1 = __builtin_amdgcn_wmma_f32_16x16x32_bf16(
                false, A1.v, false, B.v, (short)0, acc1, false, false);
        }
        float sa1 = a1[n], sc1 = c1[n], sa2 = a2[n], sc2 = cc2[n];
        for (int r = 0; r < 8; ++r) {
            int m = lhalf ? (r + 8) : r;
            size_t row0 = (size_t)(r0 + m) * DIM;
            size_t row1 = (size_t)(r0 + 16 + m) * DIM;
            float xv0 = sa1 * h[row0 + n] + sc1;      // recompute x (BN1'd residual input)
            float xv1 = sa1 * h[row1 + n] + sc1;
            out[row0 + n] = (xv0 + acc0[r]) * sa2 + sc2;
            out[row1 + n] = (xv1 + acc1[r]) * sa2 + sc2;
        }
    }
}

extern "C" void kernel_launch(void* const* d_in, const int* in_sizes, int n_in,
                              void* d_out, int out_size, void* d_ws, size_t ws_size,
                              hipStream_t stream) {
    // setup_inputs order:
    // 0:A 1:h 2:qw 3:qb 4:kw 5:kb 6:vw 7:vb 8:ow 9:ob 10:f1w 11:f1b 12:f2w 13:f2b
    // 14:bn1_g 15:bn1_b 16:bn1_m 17:bn1_v 18:bn2_g 19:bn2_b 20:bn2_m 21:bn2_v
    const float* h   = (const float*)d_in[1];
    const float* vw  = (const float*)d_in[6];
    const float* vb  = (const float*)d_in[7];
    const float* ow  = (const float*)d_in[8];
    const float* ob  = (const float*)d_in[9];
    const float* f1w = (const float*)d_in[10];
    const float* f1b = (const float*)d_in[11];
    const float* f2w = (const float*)d_in[12];
    const float* f2b = (const float*)d_in[13];
    const float* g1  = (const float*)d_in[14];
    const float* b1  = (const float*)d_in[15];
    const float* m1  = (const float*)d_in[16];
    const float* v1  = (const float*)d_in[17];
    const float* g2  = (const float*)d_in[18];
    const float* b2  = (const float*)d_in[19];
    const float* m2  = (const float*)d_in[20];
    const float* v2  = (const float*)d_in[21];

    char* ws = (char*)d_ws;
    float*  csum = (float*)ws;                               // 2 KB
    float*  cvec = (float*)(ws + 2048);                      // 8 KB (a1|c1|a2|cc2)
    __bf16* f1wT = (__bf16*)(ws + 16384);                    // 1 MB
    __bf16* f2wT = (__bf16*)(ws + 16384 + 1048576);          // 1 MB
    float*  out  = (float*)d_out;

    hipMemsetAsync(csum, 0, 512 * sizeof(float), stream);
    colsum_kernel<<<64, 256, 0, stream>>>(h, csum);
    prep_w_kernel<<<2048, 256, 0, stream>>>(f1w, f1wT, HID, DIM);   // dst[n*512+k]
    prep_w_kernel<<<2048, 256, 0, stream>>>(f2w, f2wT, DIM, HID);   // dst[n*1024+k]
    stats_kernel<<<1, 512, 0, stream>>>(csum, vw, vb, ow, ob, f2b,
                                        g1, b1, m1, v1, g2, b2, m2, v2, cvec);
    ffn_kernel<<<NROWS / MROWS, 256, 0, stream>>>(h, f1b, cvec, f1wT, f2wT, out);
}
